// the_modell_62448824483904
// MI455X (gfx1250) — matrix operand order repaired and checked by reading, not compiled, not run on hardware
//
#include <hip/hip_runtime.h>

// ---------------------------------------------------------------------------
// Problem constants (match reference)
// ---------------------------------------------------------------------------
#define NNODE  30000
#define NFEAT  512
#define NHID   64
#define NHEADS 8
#define NCLASS 500
#define NCPAD  512          // out_W columns padded to 512 for branchless staging
#define ALPHA  0.2f

typedef unsigned short u16;
typedef __attribute__((ext_vector_type(16))) __bf16 v16bf;
typedef __attribute__((ext_vector_type(8)))  float  v8f;

union Frag {
    v16bf v;
    uint4 q[2];
};

__device__ __forceinline__ u16 f2bf(float f) {
    union { float f; unsigned u; } v; v.f = f;
    unsigned r = v.u + 0x7FFFu + ((v.u >> 16) & 1u);   // round-to-nearest-even
    return (u16)(r >> 16);
}

__device__ __forceinline__ float elu1(float x) {
    return x > 0.f ? x : (__expf(x) - 1.f);
}

// ---------------------------------------------------------------------------
// Utility kernels
// ---------------------------------------------------------------------------
__global__ __launch_bounds__(256) void cvt_f32_bf16(const float* __restrict__ in,
                                                    u16* __restrict__ out, long n) {
    long i = (long)blockIdx.x * blockDim.x + threadIdx.x;
    if (i < n) out[i] = f2bf(in[i]);
}

// Transpose-convert one batch of weights:
//   out[b][n][k] = bf16(in[b][k][n]), n in [0,Npad) with zero fill for n >= Nn.
// in: [batch][K][Nn] f32   out: [batch][Npad][K] bf16
__global__ __launch_bounds__(256)
void cvt_transpose_bf16(const float* __restrict__ in, u16* __restrict__ out,
                        int K, int Nn, int Npad, long totalOut) {
    long i = (long)blockIdx.x * blockDim.x + threadIdx.x;   // over out elements
    if (i >= totalOut) return;
    int k = (int)(i % K);
    long t = i / K;
    int n = (int)(t % Npad);
    int b = (int)(t / Npad);
    u16 v = 0;
    if (n < Nn) v = f2bf(in[((long)b * K + k) * Nn + n]);
    out[i] = v;
}

__global__ __launch_bounds__(256) void fill0(float* __restrict__ p, long n) {
    long i = (long)blockIdx.x * blockDim.x + threadIdx.x;
    if (i < n) p[i] = 0.f;
}

// ---------------------------------------------------------------------------
// bf16 WMMA GEMM:  C[M x Nn] (f32, ldc=Nn) = A[M x K] (bf16, row-major)
//                                          x Bt[Npad x K] (bf16, PRE-TRANSPOSED)
// 128x64 block tile, BK=32, 128 threads = 4 waves; wave w owns rows
// [32w, 32w+32) as two 16-row A fragments; 8 WMMAs per wave per K-step.
// Staging uses CDNA5 async global->LDS loads (ASYNCcnt) with LDS double
// buffering: the loads for step i+1 fly underneath the WMMAs of step i;
// one s_wait_asynccnt + one barrier per K iteration.
// grid.z batches over Bt/C (strideB / strideC elements). K multiple of 32.
// ---------------------------------------------------------------------------
#define BM 128
#define BN 64
#define BK 32
#define KPAD 8
#define KST (BK + KPAD)             // 40 halves = 80B row stride (16B aligned)
#define SBUF ((BM + BN) * KST)      // halves per LDS buffer (A then B)

__global__ __launch_bounds__(128)
void wmma_gemm_bf16(const u16* __restrict__ A, const u16* __restrict__ Bt,
                    float* __restrict__ C, int M, int Nn, int K,
                    long strideB, long strideC)
{
    __shared__ u16 smem[2 * SBUF];   // [buf][ A: BM x KST | B: BN x KST ]

    // Escape smem's address so the compiler must assume the async-load asm
    // (which carries a "memory" clobber) writes into it; otherwise the LDS
    // fragment loads below get folded to undef (observed in round-3 asm).
    asm volatile("" :: "v"(&smem[0]) : "memory");

    const int tid  = threadIdx.x;
    const int lane = tid & 31;
    const int wave = tid >> 5;       // 0..3
    const int half = lane >> 4;      // 0/1
    const int lm   = lane & 15;

    const int mbase = blockIdx.y * BM;
    const int nbase = blockIdx.x * BN;
    const u16* Bg = Bt + (long)blockIdx.z * strideB;
    float*     Cg = C  + (long)blockIdx.z * strideC;

    // staging assignments (fixed per thread): 16B chunks
    const int ar[4] = { (tid + 0)   >> 2, (tid + 128) >> 2,
                        (tid + 256) >> 2, (tid + 384) >> 2 };  // rows 0..127
    const int ac8   = (tid & 3) * 8;                           // 0,8,16,24
    const int bn2[2] = { (tid + 0) >> 2, (tid + 128) >> 2 };   // n 0..63

    // Clamp A rows (only last M tile overhangs; garbage rows are never stored)
    int agr[4];
    #pragma unroll
    for (int it = 0; it < 4; ++it) {
        int g = mbase + ar[it];
        agr[it] = g < M ? g : (M - 1);
    }

    // async global -> LDS stage of one K-slice into buffer `buf`
    auto stage = [&](int buf, int kb) {
        #pragma unroll
        for (int it = 0; it < 4; ++it) {
            unsigned lds = (unsigned)((buf * SBUF + ar[it] * KST + ac8) * 2);
            unsigned long long ga =
                (unsigned long long)(A + (long)agr[it] * K + kb + ac8);
            asm volatile("global_load_async_to_lds_b128 %0, %1, off"
                         :: "v"(lds), "v"(ga), "v"(&smem[0]) : "memory");
        }
        #pragma unroll
        for (int it = 0; it < 2; ++it) {
            unsigned lds = (unsigned)((buf * SBUF + BM * KST + bn2[it] * KST + ac8) * 2);
            unsigned long long ga =
                (unsigned long long)(Bg + (long)(nbase + bn2[it]) * K + kb + ac8);
            asm volatile("global_load_async_to_lds_b128 %0, %1, off"
                         :: "v"(lds), "v"(ga), "v"(&smem[0]) : "memory");
        }
    };

    v8f acc[2][4] = {};

    stage(0, 0);
    asm volatile("s_wait_asynccnt 0x0" ::: "memory");
    __syncthreads();

    const int nk = K / BK;
    for (int i = 0; i < nk; ++i) {
        const int buf = i & 1;
        if (i + 1 < nk) stage(buf ^ 1, (i + 1) * BK);   // flies under the WMMAs

        // ---- A fragments (16x32): lane<16 -> M=lm, K={0..7,16..23}; lane>=16 -> K+8
        const u16* abase = &smem[buf * SBUF];
        Frag fa0, fa1;
        {
            const u16* arow = &abase[(wave * 32 + lm) * KST];
            fa0.q[0] = *(const uint4*)&arow[half * 8];
            fa0.q[1] = *(const uint4*)&arow[16 + half * 8];
            const u16* arow2 = &abase[(wave * 32 + 16 + lm) * KST];
            fa1.q[0] = *(const uint4*)&arow2[half * 8];
            fa1.q[1] = *(const uint4*)&arow2[16 + half * 8];
        }
        // ---- 4 column sub-tiles: B fragment (32x16): lane<16 -> N=lm, K={0..7,16..23}; lane>=16 -> K+8
        const u16* bbase = &smem[buf * SBUF + BM * KST];
        #pragma unroll
        for (int cs = 0; cs < 4; ++cs) {
            Frag fb;
            const u16* brow = &bbase[(cs * 16 + lm) * KST];
            fb.q[0] = *(const uint4*)&brow[half * 8];
            fb.q[1] = *(const uint4*)&brow[16 + half * 8];
            acc[0][cs] = __builtin_amdgcn_wmma_f32_16x16x32_bf16(
                false, fa0.v, false, fb.v, (short)0, acc[0][cs], false, false);
            acc[1][cs] = __builtin_amdgcn_wmma_f32_16x16x32_bf16(
                false, fa1.v, false, fb.v, (short)0, acc[1][cs], false, false);
        }
        if (i + 1 < nk) {
            asm volatile("s_wait_asynccnt 0x0" ::: "memory");
            __syncthreads();
        }
    }

    // ---- store C: VGPR r -> M = r + 8*half, N = lm
    const int mofs = half * 8;
    #pragma unroll
    for (int g = 0; g < 2; ++g) {
        #pragma unroll
        for (int cs = 0; cs < 4; ++cs) {
            int col = nbase + cs * 16 + lm;
            if (col >= Nn) continue;
            #pragma unroll
            for (int r = 0; r < 8; ++r) {
                int row = mbase + wave * 32 + g * 16 + mofs + r;
                if (row < M) Cg[(long)row * Nn + col] = acc[g][cs][r];
            }
        }
    }
}

// ---------------------------------------------------------------------------
// Per-node attention scores: s_src[n] = h[n,:].a[:F], s_dst[n] = h[n,:].a[F:]
// One wave per node; grid.z batches heads.
// ---------------------------------------------------------------------------
__global__ __launch_bounds__(256)
void score_kernel(const float* __restrict__ h, const float* __restrict__ a,
                  float* __restrict__ ssrc, float* __restrict__ sdst,
                  int Nn, int F, long strideH, long strideA, long strideS)
{
    int wave = threadIdx.x >> 5, lane = threadIdx.x & 31;
    int n = blockIdx.x * (blockDim.x >> 5) + wave;
    if (n >= Nn) return;
    const float* hh = h + (long)blockIdx.z * strideH + (long)n * F;
    const float* aa = a + (long)blockIdx.z * strideA;
    float s0 = 0.f, s1 = 0.f;
    for (int f = lane; f < F; f += 32) {
        float v = hh[f];
        s0 += v * aa[f];
        s1 += v * aa[F + f];
    }
    #pragma unroll
    for (int off = 16; off; off >>= 1) {
        s0 += __shfl_xor(s0, off, 32);
        s1 += __shfl_xor(s1, off, 32);
    }
    if (lane == 0) {
        ssrc[(long)blockIdx.z * strideS + n] = s0;
        sdst[(long)blockIdx.z * strideS + n] = s1;
    }
}

// ---------------------------------------------------------------------------
// Per-edge attention + scatter: ee = exp(-leakyrelu(s_src[s]+s_dst[d]))
//   rowsum[s] += ee;  hp[s,:] += ee * h[d,:]
// One wave per edge; grid.z batches heads. Atomics land in L2 (accumulators
// fit in 192MB L2 on MI455X).
// ---------------------------------------------------------------------------
__global__ __launch_bounds__(256)
void edge_scatter_kernel(const int* __restrict__ src, const int* __restrict__ dst,
                         const float* __restrict__ ssrc, const float* __restrict__ sdst,
                         const float* __restrict__ h, float* __restrict__ hp,
                         float* __restrict__ rowsum,
                         int E, int F, long strideH, long strideS, long strideR)
{
    int wave = threadIdx.x >> 5, lane = threadIdx.x & 31;
    int e = blockIdx.x * (blockDim.x >> 5) + wave;
    if (e >= E) return;
    int z = blockIdx.z;
    int s = src[e], d = dst[e];
    float logit = ssrc[(long)z * strideS + s] + sdst[(long)z * strideS + d];
    float lr = logit > 0.f ? logit : ALPHA * logit;
    float ee = __expf(-lr);
    if (lane == 0) atomicAdd(&rowsum[(long)z * strideR + s], ee);
    const float* hd  = h  + (long)z * strideH + (long)d * F;
    float*       hps = hp + (long)z * strideH + (long)s * F;
    for (int f = lane; f < F; f += 32)
        atomicAdd(&hps[f], ee * hd[f]);
}

// ---------------------------------------------------------------------------
// Layer-1 finalize: x_bf16[n][head*64+f] = bf16(elu(hp[head][n][f]/rowsum[head][n]))
// ---------------------------------------------------------------------------
__global__ __launch_bounds__(256)
void norm_concat_kernel(const float* __restrict__ hp, const float* __restrict__ rowsum,
                        u16* __restrict__ xb, int Nn, int F, int H)
{
    long i = (long)blockIdx.x * blockDim.x + threadIdx.x;
    long total = (long)H * Nn * F;
    if (i >= total) return;
    int f = (int)(i % F);
    long t = i / F;
    int n  = (int)(t % Nn);
    int hd = (int)(t / Nn);
    float v = hp[i] / rowsum[(long)hd * Nn + n];
    xb[(long)n * (F * H) + hd * F + f] = f2bf(elu1(v));
}

// ---------------------------------------------------------------------------
// Final: out[n][c] = elu(out[n][c] / rowsum2[n])   (in place on d_out)
// ---------------------------------------------------------------------------
__global__ __launch_bounds__(256)
void final_kernel(float* __restrict__ out, const float* __restrict__ rowsum2,
                  int Nn, int C)
{
    long i = (long)blockIdx.x * blockDim.x + threadIdx.x;
    if (i >= (long)Nn * C) return;
    int n = (int)(i / C);
    out[i] = elu1(out[i] / rowsum2[n]);
}

// ---------------------------------------------------------------------------
// Host orchestration
// ---------------------------------------------------------------------------
static inline long cdiv(long a, long b) { return (a + b - 1) / b; }

extern "C" void kernel_launch(void* const* d_in, const int* in_sizes, int n_in,
                              void* d_out, int out_size, void* d_ws, size_t ws_size,
                              hipStream_t stream) {
    const float* features = (const float*)d_in[0];
    const int*   edges    = (const int*)d_in[1];
    const float* W        = (const float*)d_in[2];
    const float* a        = (const float*)d_in[3];
    const float* out_W    = (const float*)d_in[4];
    const float* out_a    = (const float*)d_in[5];
    float* out = (float*)d_out;

    const int E = in_sizes[1] / 2;
    const int* src = edges;
    const int* dst = edges + E;

    // ---- workspace carve-up (256B aligned), with aggressive reuse
    char* ws = (char*)d_ws;
    size_t o = 0;
    auto carve = [&](size_t bytes) -> char* {
        char* p = ws + o;
        o += (bytes + 255) & ~(size_t)255;
        return p;
    };
    u16*   Xb      = (u16*)  carve((size_t)NNODE * NFEAT * 2);         // reused as xb
    u16*   Wt      = (u16*)  carve((size_t)NHEADS * NHID * NFEAT * 2); // transposed
    u16*   OWt     = (u16*)  carve((size_t)NCPAD * NFEAT * 2);         // transposed+padded
    float* h_all   = (float*)carve((size_t)NHEADS * NNODE * NHID * 4); // reused as h2
    float* hp_all  = (float*)carve((size_t)NHEADS * NNODE * NHID * 4);
    float* ssrcA   = (float*)carve((size_t)NHEADS * NNODE * 4);
    float* sdstA   = (float*)carve((size_t)NHEADS * NNODE * 4);
    float* rowsumA = (float*)carve((size_t)NHEADS * NNODE * 4);
    float* ssrc2   = (float*)carve((size_t)NNODE * 4);
    float* sdst2   = (float*)carve((size_t)NNODE * 4);
    float* rowsum2 = (float*)carve((size_t)NNODE * 4);
    (void)ws_size; (void)n_in; (void)out_size;

    const long nX   = (long)NNODE * NFEAT;          // 15.36M
    const long nWt  = (long)NHEADS * NHID * NFEAT;  // 262144
    const long nOWt = (long)NCPAD * NFEAT;          // 262144
    const long nHP  = (long)NHEADS * NNODE * NHID;  // 15.36M
    const long nO   = (long)NNODE * NCLASS;         // 15M

    // 1) bf16 conversions (weights transposed so GEMM staging is branchless)
    cvt_f32_bf16<<<dim3(cdiv(nX, 256)), 256, 0, stream>>>(features, Xb, nX);
    cvt_transpose_bf16<<<dim3(cdiv(nWt, 256)), 256, 0, stream>>>(
        W, Wt, NFEAT, NHID, NHID, nWt);
    cvt_transpose_bf16<<<dim3(cdiv(nOWt, 256)), 256, 0, stream>>>(
        out_W, OWt, NFEAT, NCLASS, NCPAD, nOWt);

    // 2) zero accumulators (fresh every call; graph-capture safe)
    fill0<<<dim3(cdiv(nHP, 256)), 256, 0, stream>>>(hp_all, nHP);
    fill0<<<dim3(cdiv((long)NHEADS * NNODE, 256)), 256, 0, stream>>>(rowsumA, (long)NHEADS * NNODE);
    fill0<<<dim3(cdiv(nO, 256)), 256, 0, stream>>>(out, nO);
    fill0<<<dim3(cdiv((long)NNODE, 256)), 256, 0, stream>>>(rowsum2, (long)NNODE);

    // 3) layer-1 GEMMs: h[head] = X @ W[head], batched over heads via grid.z
    {
        dim3 grid(cdiv(NHID, BN), cdiv(NNODE, BM), NHEADS);
        wmma_gemm_bf16<<<grid, 128, 0, stream>>>(
            Xb, Wt, h_all, NNODE, NHID, NFEAT,
            (long)NHID * NFEAT, (long)NNODE * NHID);
    }

    // 4) per-node scores (wave per node)
    score_kernel<<<dim3(cdiv(NNODE, 8), 1, NHEADS), 256, 0, stream>>>(
        h_all, a, ssrcA, sdstA, NNODE, NHID,
        (long)NNODE * NHID, (long)(2 * NHID), (long)NNODE);

    // 5) edge scatter (wave per edge)
    edge_scatter_kernel<<<dim3(cdiv(E, 8), 1, NHEADS), 256, 0, stream>>>(
        src, dst, ssrcA, sdstA, h_all, hp_all, rowsumA,
        E, NHID, (long)NNODE * NHID, (long)NNODE, (long)NNODE);

    // 6) normalize + ELU + concat heads -> bf16 x (reuses Xb buffer)
    norm_concat_kernel<<<dim3(cdiv(nHP, 256)), 256, 0, stream>>>(
        hp_all, rowsumA, Xb, NNODE, NHID, NHEADS);

    // 7) output GEMM: h2 = x @ out_W  (h2 reuses h_all buffer)
    float* h2 = h_all;
    {
        dim3 grid(cdiv(NCLASS, BN), cdiv(NNODE, BM), 1);
        wmma_gemm_bf16<<<grid, 128, 0, stream>>>(
            Xb, OWt, h2, NNODE, NCLASS, NHID * NHEADS, 0L, 0L);
    }

    // 8) output scores
    score_kernel<<<dim3(cdiv(NNODE, 8), 1, 1), 256, 0, stream>>>(
        h2, out_a, ssrc2, sdst2, NNODE, NCLASS, 0L, 0L, 0L);

    // 9) output edge scatter, accumulating directly into d_out
    edge_scatter_kernel<<<dim3(cdiv(E, 8), 1, 1), 256, 0, stream>>>(
        src, dst, ssrc2, sdst2, h2, out, rowsum2,
        E, NCLASS, 0L, 0L, 0L);

    // 10) final normalize + ELU in place
    final_kernel<<<dim3(cdiv(nO, 256)), 256, 0, stream>>>(out, rowsum2, NNODE, NCLASS);
}
